// BiDirectionalSpatialMamba_33483565039900
// MI455X (gfx1250) — compile-verified, matrix operand order checked by reading, not run on hardware
//
#include <hip/hip_runtime.h>
#include <cstddef>
#include <cstdint>

typedef __attribute__((ext_vector_type(16))) __bf16 v16bf;
typedef __attribute__((ext_vector_type(8)))  float  v8f;
typedef int v4i_ __attribute__((vector_size(16)));

#define HID   256
#define DIN   512
#define G3    768
#define PLEN  1024
#define MROWS 131072   // (8*16)*1024 flattened rows

// ---- CDNA5 async global->LDS copy (ASYNCcnt path), guarded ---------------
#if defined(__has_builtin)
#if __has_builtin(__builtin_amdgcn_global_load_async_to_lds_b128) && \
    __has_builtin(__builtin_amdgcn_s_wait_asynccnt)
#define USE_ASYNC_LDS 1
#endif
#endif

__device__ __forceinline__ void async_g2l_b128(const void* g, void* l) {
#ifdef USE_ASYNC_LDS
  __builtin_amdgcn_global_load_async_to_lds_b128(
      (__attribute__((address_space(1))) v4i_*)(void*)g,
      (__attribute__((address_space(3))) v4i_*)l, 0, 0);
#else
  __builtin_memcpy(l, g, 16);
#endif
}
__device__ __forceinline__ void async_wait0() {
#ifdef USE_ASYNC_LDS
  __builtin_amdgcn_s_wait_asynccnt(0);
#endif
}

// ---------------- WMMA fragment index math (v_wmma_f32_16x16x32_bf16) -----
__device__ __forceinline__ int frag_k(int e, int half) {
  int vg = e >> 1, w = e & 1;
  int base = (vg < 4) ? (vg * 2) : (16 + (vg - 4) * 2);
  return base + w + half * 8;
}
// inverse: K (0..31) -> (half, element)
__device__ __forceinline__ void k_to_le(int k, int& half, int& e) {
  half = (k >> 3) & 1;
  int kp = k - half * 8;  // {0..7} U {16..23}
  int vg = (kp < 8) ? (kp >> 1) : (4 + ((kp - 16) >> 1));
  e = vg * 2 + (kp & 1);
}

__device__ __forceinline__ v8f wmma_bf16(v16bf a, v16bf b, v8f c) {
  return __builtin_amdgcn_wmma_f32_16x16x32_bf16(false, a, false, b, (short)0,
                                                 c, false, false);
}

// ---------------- fragment-major weight pre-pack --------------------------
// pk[((kt*(N/16) + nt)*32 + lane)*16 + e]: each fragment is one aligned 32B
// chunk per lane; a (BK x BN) tile is a CONTIGUOUS 8KB block -> async-copy.
__global__ void pack_frag(const float* __restrict__ W, __bf16* __restrict__ pk,
                          int K, int N) {
  const int g = blockIdx.x * blockDim.x + threadIdx.x;
  const int lane = g & 31, fid = g >> 5;
  if (fid >= (K >> 5) * (N >> 4)) return;
  const int ntAll = N >> 4;
  const int kt = fid / ntAll, nt = fid % ntAll;
  const int col = lane & 15, half = lane >> 4;
  v16bf f;
#pragma unroll
  for (int e = 0; e < 16; ++e)
    f[e] = (__bf16)W[(size_t)(kt * 32 + frag_k(e, half)) * N + nt * 16 + col];
  *(v16bf*)(pk + ((size_t)fid * 32 + lane) * 16) = f;
}

// ---------------- generic bf16 WMMA GEMM, 64x128 block tile ---------------
// A staged (and f32->bf16 converted) into frag-packed LDS by threads;
// B tile (pre-packed global) streamed linearly with async global->LDS.
#define BM 64
#define BN 128
#define BK 32

// EPI: 0 = +bias, exact GELU, store bf16
//      1 = +bias, store bf16
//      2 = +bias +fp32 residual, store fp32
template <int AF32, int EPI>
__global__ __launch_bounds__(256) void wmma_gemm(
    const void* __restrict__ Av, const __bf16* __restrict__ Bpk,
    const float* __restrict__ bias, const float* __restrict__ resid,
    void* __restrict__ Cv, int M, int N, int K) {
  __shared__ __align__(32) __bf16 As[4][32][16];  // 4 M-tiles, frag-packed
  __shared__ __align__(32) __bf16 Bs[8][32][16];  // 8 N-tiles, frag-packed

  const int tid = threadIdx.x;
  const int wave = tid >> 5, lane = tid & 31;
  const int wm = wave & 3;   // M sub-tile (4 x 16)
  const int wn = wave >> 2;  // N half (2 x 64)
  const long rowBlk = (long)blockIdx.y * BM;
  const long colBlk = (long)blockIdx.x * BN;

  const int ar = tid >> 2, ac0 = (tid & 3) * 8;  // A staging: 8 elems/thread
  const int amt = ar >> 4, arow = ar & 15;

  v8f acc[4];
#pragma unroll
  for (int t = 0; t < 4; ++t)
#pragma unroll
    for (int i = 0; i < 8; ++i) acc[t][i] = 0.0f;

  for (int k0 = 0; k0 < K; k0 += BK) {
    __syncthreads();
    // B tile: contiguous 8KB fragment block, async global->LDS (2x b128/thr)
    {
      const char* gsrc =
          (const char*)(Bpk +
                        ((size_t)(k0 >> 5) * (N >> 4) + (colBlk >> 4)) * 512) +
          tid * 32;
      char* ldst = (char*)&Bs[0][0][0] + tid * 32;
      async_g2l_b128(gsrc, ldst);
      async_g2l_b128(gsrc + 16, ldst + 16);
    }
    // A tile: load (convert) + frag-pack into LDS
    if (AF32) {
      const float* ap = (const float*)Av + (rowBlk + ar) * K + k0 + ac0;
      __builtin_prefetch(ap + BK, 0, 1);  // global_prefetch_b8
#pragma unroll
      for (int i = 0; i < 8; ++i) {
        int half, e;
        k_to_le(ac0 + i, half, e);
        As[amt][half * 16 + arow][e] = (__bf16)ap[i];
      }
    } else {
      const __bf16* ap = (const __bf16*)Av + (rowBlk + ar) * K + k0 + ac0;
      __builtin_prefetch(ap + BK, 0, 1);
#pragma unroll
      for (int i = 0; i < 8; ++i) {
        int half, e;
        k_to_le(ac0 + i, half, e);
        As[amt][half * 16 + arow][e] = ap[i];
      }
    }
    async_wait0();
    __syncthreads();

    v16bf a = *(const v16bf*)&As[wm][lane][0];  // 2x ds_load_b128
#pragma unroll
    for (int t = 0; t < 4; ++t) {
      v16bf b = *(const v16bf*)&Bs[wn * 4 + t][lane][0];
      acc[t] = wmma_bf16(a, b, acc[t]);
    }
  }

  // C tile lane layout: n = lane&15, m = (lane>>4)*8 + r
  const int cn = lane & 15;
  const int m0 = (lane >> 4) * 8;
#pragma unroll
  for (int t = 0; t < 4; ++t) {
    const long gc = colBlk + wn * 64 + t * 16 + cn;
#pragma unroll
    for (int r = 0; r < 8; ++r) {
      const long gr = rowBlk + wm * 16 + m0 + r;
      float v = acc[t][r] + bias[gc];
      if (EPI == 0) {
        v = 0.5f * v * (1.0f + erff(v * 0.70710678118f));  // exact GELU
        ((__bf16*)Cv)[gr * N + gc] = (__bf16)v;
      } else if (EPI == 1) {
        ((__bf16*)Cv)[gr * N + gc] = (__bf16)v;
      } else {
        ((float*)Cv)[gr * N + gc] = v + resid[gr * N + gc];
      }
    }
  }
}

// ---------------- GRU scan: 8 WGs/branch, 16 batch rows each --------------
// 512 threads = 16 waves; wave w owns gate columns [48w, 48w+48).
// fp32 state in thread registers; bf16 frag-packed h copy in LDS feeds the
// WMMA A operand; W_hh register-resident (frag-major, hoisted by RA).
// kt-outer (transient A-frag, 3 accs) + contiguous b128 h writeback.
__global__ __launch_bounds__(512) void gru_scan(
    const __bf16* __restrict__ gi,   // (MROWS, 768), row m = n*1024 + p
    const __bf16* __restrict__ Wpk,  // fragment-packed W_hh (bf16)
    const float* __restrict__ bhh,   // (768)
    __bf16* __restrict__ y,          // (MROWS, 512)
    int dir, int colOff) {
  __shared__ __align__(32) __bf16 hpk[8][32][16];  // 8 KB bf16 h, frag-packed
  __shared__ float ghS[16][G3];                    // 48 KB gate pre-acts

  const int tid = threadIdx.x;
  const int wave = tid >> 5, lane = tid & 31;
  const long n0 = (long)blockIdx.x * 16;

  for (int i = tid; i < 8 * 32 * 16; i += 512) ((__bf16*)hpk)[i] = (__bf16)0.0f;

  // per-thread persistent state: row = tid>>5 (0..15), cols j0..j0+7.
  // These 8 elements are CONTIGUOUS in hpk (same kt, same half, e = e0+jj)
  // because j0 is 8-aligned -> single b128 writeback at byte offset hbase.
  const int row = tid >> 5;
  const int j0 = (tid & 31) * 8;
  float hreg[8];
#pragma unroll
  for (int jj = 0; jj < 8; ++jj) hreg[jj] = 0.0f;
  int hbase;
  {
    int half, e;
    k_to_le(j0 & 31, half, e);
    hbase = (((j0 >> 5) * 32 + half * 16 + row) * 16 + e) * 2;
  }

  // this (wave, lane)'s gh columns are fixed: hoist the 3 b_hh values
  const int cn = lane & 15, m0 = (lane >> 4) * 8;
  float bh3[3];
#pragma unroll
  for (int nt = 0; nt < 3; ++nt) bh3[nt] = bhh[wave * 48 + nt * 16 + cn];

  const __bf16* Wb = Wpk + ((size_t)(wave * 3) * 32 + lane) * 16;
  const int pstep = dir ? -1 : 1;
  int p = dir ? (PLEN - 1) : 0;  // output position == input position
  __syncthreads();

  for (int t = 0; t < PLEN; ++t, p += pstep) {
    // phase 1: gh = h @ W_hh (16 x 768); kt-outer: A-frag transient,
    // 3 accumulators live, 24 WMMA per wave per step
    v8f acc[3];
#pragma unroll
    for (int nt = 0; nt < 3; ++nt)
#pragma unroll
      for (int i = 0; i < 8; ++i) acc[nt][i] = 0.0f;
#pragma unroll
    for (int kt = 0; kt < 8; ++kt) {
      v16bf a = *(const v16bf*)&hpk[kt][lane][0];  // 2x ds_load_b128
#pragma unroll
      for (int nt = 0; nt < 3; ++nt) {
        v16bf b = *(const v16bf*)(Wb + (size_t)(kt * 48 + nt) * 32 * 16);
        acc[nt] = wmma_bf16(a, b, acc[nt]);
      }
    }
#pragma unroll
    for (int nt = 0; nt < 3; ++nt)
#pragma unroll
      for (int r = 0; r < 8; ++r)
        ghS[m0 + r][wave * 48 + nt * 16 + cn] = acc[nt][r] + bh3[nt];
    __syncthreads();

    // phase 2: gates + state update (512 thr x 8 = 16 rows x 256)
    {
      const long m = (n0 + row) * (long)PLEN + p;
      const __bf16* gir = gi + m * G3 + j0;
      __bf16* yr = y + m * (2 * HID) + colOff + j0;
      if (t + 1 < PLEN) {  // prefetch next step's input-gate row
        const __bf16* gn = gir + (long)pstep * G3;
        __builtin_prefetch(gn, 0, 1);
        __builtin_prefetch(gn + HID, 0, 1);
        __builtin_prefetch(gn + 2 * HID, 0, 1);
      }
      __bf16 gR[8], gZ[8], gN[8], hO[8];
      __builtin_memcpy(gR, gir, 16);  // b128 loads
      __builtin_memcpy(gZ, gir + HID, 16);
      __builtin_memcpy(gN, gir + 2 * HID, 16);
#pragma unroll
      for (int jj = 0; jj < 8; ++jj) {
        const int j = j0 + jj;
        const float hr = ghS[row][j];
        const float hz = ghS[row][j + HID];
        const float hn = ghS[row][j + 2 * HID];
        const float rg = 1.0f / (1.0f + __expf(-((float)gR[jj] + hr)));
        const float zg = 1.0f / (1.0f + __expf(-((float)gZ[jj] + hz)));
        const float ng = tanhf((float)gN[jj] + rg * hn);
        const float hv = (1.0f - zg) * ng + zg * hreg[jj];
        hreg[jj] = hv;
        hO[jj] = (__bf16)hv;
      }
      __builtin_memcpy((char*)&hpk[0][0][0] + hbase, hO, 16);  // b128 ds store
      __builtin_memcpy(yr, hO, 16);                            // b128 store
    }
    __syncthreads();
  }
}

// ---------------- LayerNorm (one wave32 per 256-wide row) -----------------
__global__ __launch_bounds__(256) void ln_rows(const float* __restrict__ pre,
                                               const float* __restrict__ g,
                                               const float* __restrict__ b,
                                               float* __restrict__ out) {
  const long row = (long)blockIdx.x * 8 + (threadIdx.x >> 5);
  const int lane = threadIdx.x & 31;
  const float* pr = pre + row * HID;
  float v[8], s = 0.f, s2 = 0.f;
#pragma unroll
  for (int i = 0; i < 8; ++i) {
    v[i] = pr[lane + i * 32];
    s += v[i];
    s2 += v[i] * v[i];
  }
#pragma unroll
  for (int off = 16; off > 0; off >>= 1) {
    s += __shfl_xor(s, off, 32);
    s2 += __shfl_xor(s2, off, 32);
  }
  const float mu = s * (1.0f / HID);
  const float var = s2 * (1.0f / HID) - mu * mu;
  const float rs = rsqrtf(var + 1e-5f);
  float* po = out + row * HID;
#pragma unroll
  for (int i = 0; i < 8; ++i) {
    const int c = lane + i * 32;
    po[c] = (v[i] - mu) * rs * g[c] + b[c];
  }
}

// ---------------- host launch ---------------------------------------------
extern "C" void kernel_launch(void* const* d_in, const int* in_sizes, int n_in,
                              void* d_out, int out_size, void* d_ws,
                              size_t ws_size, hipStream_t stream) {
  const float* x = (const float*)d_in[0];
  const float* lin_W[2] = {(const float*)d_in[1], (const float*)d_in[7]};
  const float* lin_b[2] = {(const float*)d_in[2], (const float*)d_in[8]};
  const float* W_ih[2] = {(const float*)d_in[3], (const float*)d_in[9]};
  const float* W_hh[2] = {(const float*)d_in[4], (const float*)d_in[10]};
  const float* b_ih[2] = {(const float*)d_in[5], (const float*)d_in[11]};
  const float* b_hh[2] = {(const float*)d_in[6], (const float*)d_in[12]};
  const float* proj_W = (const float*)d_in[13];
  const float* proj_b = (const float*)d_in[14];
  const float* ln_g = (const float*)d_in[15];
  const float* ln_b = (const float*)d_in[16];

  char* ws = (char*)d_ws;
  size_t off = 0;
  auto take = [&](size_t bytes) -> void* {
    void* p = ws + off;
    off = (off + bytes + 255) & ~(size_t)255;
    return p;
  };

  __bf16 *linpk[2], *ihpk[2];
  for (int s = 0; s < 2; ++s) {
    linpk[s] = (__bf16*)take((size_t)HID * DIN * 2);
    ihpk[s] = (__bf16*)take((size_t)DIN * G3 * 2);
  }
  __bf16* projpk = (__bf16*)take((size_t)(2 * HID) * HID * 2);
  __bf16* whhpk = (__bf16*)take((size_t)HID * G3 * 2);
  __bf16* h_bf = (__bf16*)take((size_t)MROWS * DIN * 2);      // 128 MB
  __bf16* gi_bf = (__bf16*)take((size_t)MROWS * G3 * 2);      // 192 MB
  __bf16* y_bf = (__bf16*)take((size_t)MROWS * 2 * HID * 2);  // 128 MB
  float* pre = (float*)gi_bf;  // fp32 pre-LN output reuses gi region

  auto pack = [&](const float* W, __bf16* pk, int K, int N) {
    int thr = (K >> 5) * (N >> 4) * 32;
    pack_frag<<<(thr + 255) / 256, 256, 0, stream>>>(W, pk, K, N);
  };
  for (int s = 0; s < 2; ++s) {
    pack(lin_W[s], linpk[s], HID, DIN);
    pack(W_ih[s], ihpk[s], DIN, G3);
  }
  pack(proj_W, projpk, 2 * HID, HID);

  for (int s = 0; s < 2; ++s) {  // branch 0 = fwd, 1 = bwd (reused buffers)
    wmma_gemm<1, 0><<<dim3(DIN / BN, MROWS / BM), 256, 0, stream>>>(
        x, linpk[s], lin_b[s], nullptr, h_bf, MROWS, DIN, HID);
    wmma_gemm<0, 1><<<dim3(G3 / BN, MROWS / BM), 256, 0, stream>>>(
        h_bf, ihpk[s], b_ih[s], nullptr, gi_bf, MROWS, G3, DIN);
    pack(W_hh[s], whhpk, HID, G3);
    gru_scan<<<8, 512, 0, stream>>>(gi_bf, whhpk, b_hh[s], y_bf, s, s * HID);
  }
  wmma_gemm<0, 2><<<dim3(HID / BN, MROWS / BM), 256, 0, stream>>>(
      y_bf, projpk, proj_b, x, pre, MROWS, HID, 2 * HID);
  ln_rows<<<MROWS / 8, 256, 0, stream>>>(pre, ln_g, ln_b, (float*)d_out);
}
